// Head_21045339750845
// MI455X (gfx1250) — matrix operand order repaired and checked by reading, not compiled, not run on hardware
//
#include <hip/hip_runtime.h>

// ---------------------------------------------------------------------------
// CDNA5 (gfx1250) fused causal attention head, bf16 WMMA path with TDM
// (tensor_load_to_lds) staging of weights / K / V into LDS.
// x:[128,512,384] f32, Wq/Wk/Wv:[384,64] f32 -> out:[128,512,64] f32
// ---------------------------------------------------------------------------

typedef __attribute__((ext_vector_type(16))) __bf16 v16bf;
typedef __attribute__((ext_vector_type(8)))  float  v8f;
typedef __attribute__((ext_vector_type(4)))  unsigned int v4u;
typedef __attribute__((ext_vector_type(8)))  int    v8i;
typedef __attribute__((ext_vector_type(4)))  int    v4i;

union ABu {           // one WMMA A/B operand: 16 bf16 per lane = 2 x b128 load
    v16bf v;
    uint4 q[2];
};

static __device__ __forceinline__ v8f wmma_bf16(v16bf a, v16bf b, v8f c) {
    // (neg_a, A, neg_b, B, c_mod, C, reuse_a, reuse_b)
    return __builtin_amdgcn_wmma_f32_16x16x32_bf16(false, a, false, b,
                                                   (short)0, c, false, false);
}

static __device__ __forceinline__ unsigned short f2bf(float f) {
    unsigned int x = __float_as_uint(f);
    x += 0x7FFFu + ((x >> 16) & 1u);   // round-to-nearest-even
    return (unsigned short)(x >> 16);
}

// ---------------------------------------------------------------------------
// TDM: 1-D global -> LDS DMA of nbytes (multiple of 4, ndwords <= 0xFFFF).
// D# per CDNA5 ISA ch.8: group0 {count=1, lds_addr, global_addr, type=2},
// group1 {data_size=4B, tensor_dim0=tile_dim0=ndw, dims1=1, stride0=ndw},
// groups 2/3 zero. 6-arg builtin on this toolchain. All inputs WG-uniform.
// ---------------------------------------------------------------------------
static __device__ __forceinline__ void tdm_load_1d(const void* gsrc,
                                                   unsigned lds_byte_off,
                                                   unsigned nbytes) {
    unsigned ndw = nbytes >> 2;
    unsigned long long ga = (unsigned long long)(uintptr_t)gsrc;
    v4u g0 = { 1u,                                        // count=1
               lds_byte_off,                              // lds_addr
               (unsigned)(ga & 0xFFFFFFFFu),              // global_addr[31:0]
               (unsigned)((ga >> 32) & 0x1FFFFFFu) | (2u << 30) }; // [56:32]|type=2
    v8i g1 = { (int)(2u << 16),                           // data_size code 2 = 4B
               (int)((ndw & 0xFFFFu) << 16),              // tensor_dim0[15:0]
               (int)(((ndw >> 16) & 0xFFFFu) | (1u << 16)),// dim0[31:16] | tensor_dim1=1
               (int)((ndw & 0xFFFFu) << 16),              // tile_dim0
               1,                                         // tile_dim1=1
               (int)ndw,                                  // tensor_dim0_stride[31:0]
               0, 0 };
    v4i z4 = { 0, 0, 0, 0 };
    v8i z8 = { 0, 0, 0, 0, 0, 0, 0, 0 };
    __builtin_amdgcn_tensor_load_to_lds(g0, g1, z4, z4, z8, 0);
}

// sizes
#define BB 128
#define TT 512
#define CC 384
#define DD 64
#define MROWS (BB * TT)            // 65536
#define WELEM (DD * CC)            // 24576 per weight matrix

// ---------------------------------------------------------------------------
// Kernel 1: x fp32 -> bf16 (pairs)
// ---------------------------------------------------------------------------
__global__ __launch_bounds__(256) void cvt_x(const float* __restrict__ x,
                                             unsigned short* __restrict__ xb,
                                             int npairs) {
    int i = blockIdx.x * blockDim.x + threadIdx.x;
    if (i < npairs) {
        float2 f = ((const float2*)x)[i];
        unsigned int p = (unsigned int)f2bf(f.x) | ((unsigned int)f2bf(f.y) << 16);
        ((unsigned int*)xb)[i] = p;
    }
}

// ---------------------------------------------------------------------------
// Kernel 2: W[C][D] fp32 -> Wt[3][D][C] bf16 (transposed: B-tiles contiguous in K)
// ---------------------------------------------------------------------------
__global__ __launch_bounds__(256) void cvt_w(const float* __restrict__ Wq,
                                             const float* __restrict__ Wk,
                                             const float* __restrict__ Wv,
                                             unsigned short* __restrict__ wt) {
    int i = blockIdx.x * blockDim.x + threadIdx.x;
    if (i >= 3 * WELEM) return;
    int mat = i / WELEM;
    int rem = i - mat * WELEM;
    int d = rem / CC;
    int c = rem - d * CC;
    const float* W = (mat == 0) ? Wq : ((mat == 1) ? Wk : Wv);
    wt[i] = f2bf(W[c * DD + d]);
}

// ---------------------------------------------------------------------------
// Kernel 3: QKV projection. TDM stages all 144 KB of weights into LDS once
// per block; each wave computes a 16-row tile of Q,K,V (12 n-tiles).
// Per K-step: batch 24 ds_load_b128 (pure immediate offsets off two base
// pointers), sched_barrier, then 6+6 back-to-back WMMAs.
// Dynamic LDS: 147456 bytes.
// ---------------------------------------------------------------------------
__global__ __launch_bounds__(256) void qkv_kernel(const unsigned short* __restrict__ xb,
                                                  const unsigned short* __restrict__ wt,
                                                  unsigned short* __restrict__ qb,
                                                  unsigned short* __restrict__ kb,
                                                  unsigned short* __restrict__ vt) {
    extern __shared__ __align__(16) unsigned short smem_w[];   // [3*WELEM]

    if ((threadIdx.x >> 5) == 0) {           // wave 0 issues the DMA
        tdm_load_1d(wt, 0, 3 * WELEM * 2);   // 147456 B = 36864 dwords
        __builtin_amdgcn_s_wait_tensorcnt(0);
    }
    __syncthreads();

    int wave  = blockIdx.x * 8 + (threadIdx.x >> 5);   // 4096 row tiles total
    int lane  = threadIdx.x & 31;
    int lrow  = lane & 15;
    int lhalf = lane >> 4;
    int row0  = wave << 4;

    v8f acc[12];
#pragma unroll
    for (int t = 0; t < 12; t++)
#pragma unroll
        for (int r = 0; r < 8; r++) acc[t][r] = 0.0f;

    // A-layout: lane<16 holds K {0..7}|{16..23}; lane>=16 holds {8..15}|{24..31}
    const unsigned short* xrow = xb + (size_t)(row0 + lrow) * CC + lhalf * 8;
    ABu a, an;
    a.q[0] = *(const uint4*)(xrow);
    a.q[1] = *(const uint4*)(xrow + 16);

    // lane's B-column base in LDS; tile t at byte offset t*16*CC*2 (0..135168),
    // so keep a second base for tiles 6..11 to stay within DS imm offsets.
    const unsigned short* wbase = smem_w + lrow * CC + lhalf * 8;

#pragma unroll 1
    for (int kk = 0; kk < CC; kk += 32) {
        if (kk + 32 < CC) {                  // prefetch next A tile (registers)
            const unsigned short* apn = xrow + kk + 32;
            an.q[0] = *(const uint4*)(apn);
            an.q[1] = *(const uint4*)(apn + 16);
        }
        const unsigned short* wk  = wbase + kk;
        const unsigned short* wk2 = wk + 6 * 16 * CC;
        ABu bt[12];
#pragma unroll
        for (int j = 0; j < 6; j++) {        // tiles 0..5: imm offsets 0..61440
            bt[j].q[0] = *(const uint4*)(wk + j * (16 * CC));
            bt[j].q[1] = *(const uint4*)(wk + j * (16 * CC) + 16);
        }
#pragma unroll
        for (int j = 0; j < 6; j++) {        // tiles 6..11 off second base
            bt[6 + j].q[0] = *(const uint4*)(wk2 + j * (16 * CC));
            bt[6 + j].q[1] = *(const uint4*)(wk2 + j * (16 * CC) + 16);
        }
        __builtin_amdgcn_sched_barrier(0);   // keep loads above the WMMA block
#pragma unroll
        for (int t = 0; t < 12; t++)         // 12 back-to-back WMMAs
            acc[t] = wmma_bf16(a.v, bt[t].v, acc[t]);
        a = an;
    }

    // stores: C/D layout -> col d = n0+lrow, row m = lhalf*8 + r
    int bq = row0 >> 9;        // 16-row tiles never cross batch (512 % 16 == 0)
    int tq = row0 & 511;
#pragma unroll
    for (int t = 0; t < 12; t++) {
        int mat = t >> 2;
        int n0  = (t & 3) << 4;
        int d   = n0 + lrow;
        if (mat == 0) {
#pragma unroll
            for (int r = 0; r < 8; r++)
                qb[(size_t)(row0 + lhalf * 8 + r) * DD + d] = f2bf(acc[t][r]);
        } else if (mat == 1) {
#pragma unroll
            for (int r = 0; r < 8; r++)
                kb[(size_t)(row0 + lhalf * 8 + r) * DD + d] = f2bf(acc[t][r]);
        } else {
            // V transposed: vt[b][d][t], pack pairs along t
            unsigned int* vrow = (unsigned int*)(vt +
                ((size_t)(bq * DD + d) * TT + tq + lhalf * 8));
#pragma unroll
            for (int r = 0; r < 8; r += 2)
                vrow[r >> 1] = (unsigned int)f2bf(acc[t][r]) |
                               ((unsigned int)f2bf(acc[t][r + 1]) << 16);
        }
    }
}

// ---------------------------------------------------------------------------
// Kernel 4: fused causal flash attention. One wave per 16-query tile; every
// block handles 8 consecutive q-tiles of ONE batch, so the block TDM-stages
// that batch's K [512,64] and V^T [64,512] (64 KB each) into LDS.
// Dynamic LDS: 64K (K) + 64K (V^T) + 8K (P staging) = 139264 bytes.
// ---------------------------------------------------------------------------
__global__ __launch_bounds__(256) void attn_kernel(const unsigned short* __restrict__ qb,
                                                   const unsigned short* __restrict__ kb,
                                                   const unsigned short* __restrict__ vt,
                                                   float* __restrict__ out) {
    extern __shared__ __align__(16) unsigned short smem_a[];
    unsigned short* ksh = smem_a;                 // [512*64]
    unsigned short* vsh = smem_a + 32768;         // [64*512]

    int wid   = threadIdx.x >> 5;
    int lane  = threadIdx.x & 31;
    int wave  = blockIdx.x * 8 + wid;             // 4096 waves
    int b     = wave >> 5;                        // uniform per block (8 | 32)
    int q0    = (wave & 31) << 4;
    int lrow  = lane & 15;
    int lhalf = lane >> 4;

    const unsigned short* kbb = kb + (size_t)b * TT * DD;
    const unsigned short* vtb = vt + (size_t)b * DD * TT;

    if (wid == 0) {                               // wave 0 issues both DMAs
        tdm_load_1d(kbb, 0,     TT * DD * 2);     // K  -> LDS[0]
        tdm_load_1d(vtb, 65536, DD * TT * 2);     // V^T-> LDS[65536]
        __builtin_amdgcn_s_wait_tensorcnt(0);
    }
    __syncthreads();

    // Q in A-layout, both K-chunks (d 0..31, 32..63), loaded once from global
    const unsigned short* qrow =
        qb + ((size_t)(b * TT + q0 + lrow)) * DD + lhalf * 8;
    ABu a0, a1;
    a0.q[0] = *(const uint4*)(qrow);
    a0.q[1] = *(const uint4*)(qrow + 16);
    a1.q[0] = *(const uint4*)(qrow + 32);
    a1.q[1] = *(const uint4*)(qrow + 48);

    float mr[8], ls[8];
    v8f o[4];
#pragma unroll
    for (int r = 0; r < 8; r++) { mr[r] = -1e30f; ls[r] = 0.0f; }
#pragma unroll
    for (int nt = 0; nt < 4; nt++)
#pragma unroll
        for (int r = 0; r < 8; r++) o[nt][r] = 0.0f;

    const float scale = 0.05103103630798287f;     // 384^-0.5 (n_embed scaling)
    unsigned short* ps = smem_a + 65536 + (wid << 9);   // per-wave 16x32 bf16
    const unsigned short* kr0 = ksh + lrow * DD + lhalf * 8;
    const unsigned short* vr0 = vsh + lrow * TT + lhalf * 8;
    int send = q0 + 16;

#pragma unroll 1
    for (int s0 = 0; s0 < send; s0 += 32) {
        // ---- S = Q * K^T for 32 keys: preload all 4 K B-chunks, then
        //      4 WMMAs with the two accumulation chains interleaved ----
        const unsigned short* kr = kr0 + s0 * DD;
        ABu kbt[4];            // [t2*2 + dchunk]; imm offsets 0..2144 bytes
#pragma unroll
        for (int t2 = 0; t2 < 2; t2++) {
            const unsigned short* krt = kr + (t2 << 4) * DD;
            kbt[t2 * 2 + 0].q[0] = *(const uint4*)(krt);
            kbt[t2 * 2 + 0].q[1] = *(const uint4*)(krt + 16);
            kbt[t2 * 2 + 1].q[0] = *(const uint4*)(krt + 32);
            kbt[t2 * 2 + 1].q[1] = *(const uint4*)(krt + 48);
        }
        __builtin_amdgcn_sched_barrier(0);
        v8f s0a, s1a;
#pragma unroll
        for (int r = 0; r < 8; r++) { s0a[r] = 0.0f; s1a[r] = 0.0f; }
        s0a = wmma_bf16(a0.v, kbt[0].v, s0a);
        s1a = wmma_bf16(a0.v, kbt[2].v, s1a);
        s0a = wmma_bf16(a1.v, kbt[1].v, s0a);
        s1a = wmma_bf16(a1.v, kbt[3].v, s1a);

        // ---- online softmax (row m = lhalf*8+r lives in one VGPR across a
        //      16-lane half; reduce with xor shuffles 8,4,2,1) ----
#pragma unroll
        for (int r = 0; r < 8; r++) {
            int m    = lhalf * 8 + r;
            int qidx = q0 + m;
            float v0 = s0a[r] * scale;
            float v1 = s1a[r] * scale;
            if (s0 + lrow > qidx)      v0 = -1e30f;   // causal mask
            if (s0 + 16 + lrow > qidx) v1 = -1e30f;
            float tmax = fmaxf(v0, v1);
#pragma unroll
            for (int off = 8; off >= 1; off >>= 1)
                tmax = fmaxf(tmax, __shfl_xor(tmax, off, 32));
            float mn   = fmaxf(mr[r], tmax);
            float corr = __expf(mr[r] - mn);
            float p0   = __expf(v0 - mn);
            float p1   = __expf(v1 - mn);
            float psum = p0 + p1;
#pragma unroll
            for (int off = 8; off >= 1; off >>= 1)
                psum += __shfl_xor(psum, off, 32);
            ls[r] = ls[r] * corr + psum;
            mr[r] = mn;
#pragma unroll
            for (int nt = 0; nt < 4; nt++) o[nt][r] *= corr;
            // stage P row-major 16x32 bf16 (same-wave DS ops are in-order)
            ps[m * 32 + lrow]      = f2bf(p0);
            ps[m * 32 + 16 + lrow] = f2bf(p1);
        }

        // ---- reload P in A-layout ----
        const unsigned short* pr = ps + lrow * 32 + lhalf * 8;
        ABu pa;
        pa.q[0] = *(const uint4*)(pr);
        pa.q[1] = *(const uint4*)(pr + 16);

        // ---- O += P * V: preload all 4 V B-chunks, then 4 WMMAs ----
        ABu vbt[4];
#pragma unroll
        for (int nt = 0; nt < 4; nt++) {
            const unsigned short* vr = vr0 + (nt << 4) * TT + s0;
            vbt[nt].q[0] = *(const uint4*)(vr);
            vbt[nt].q[1] = *(const uint4*)(vr + 16);
        }
        __builtin_amdgcn_sched_barrier(0);
#pragma unroll
        for (int nt = 0; nt < 4; nt++)
            o[nt] = wmma_bf16(pa.v, vbt[nt].v, o[nt]);
    }

    // ---- epilogue: normalize and store fp32 ----
#pragma unroll
    for (int r = 0; r < 8; r++) {
        int m = lhalf * 8 + r;
        float inv = 1.0f / ls[r];
        float* orow = out + ((size_t)(b * TT + q0 + m)) * DD + lrow;
#pragma unroll
        for (int nt = 0; nt < 4; nt++)
            orow[nt << 4] = o[nt][r] * inv;
    }
}

// ---------------------------------------------------------------------------
// Launch
// ---------------------------------------------------------------------------
extern "C" void kernel_launch(void* const* d_in, const int* in_sizes, int n_in,
                              void* d_out, int out_size, void* d_ws, size_t ws_size,
                              hipStream_t stream) {
    const float* x  = (const float*)d_in[0];
    const float* Wq = (const float*)d_in[1];
    const float* Wk = (const float*)d_in[2];
    const float* Wv = (const float*)d_in[3];

    // workspace layout (bytes)
    char* ws = (char*)d_ws;
    unsigned short* xb = (unsigned short*)(ws);                      // 50,331,648
    unsigned short* wt = (unsigned short*)(ws + 50331648);           //    147,456
    unsigned short* qb = (unsigned short*)(ws + 50479104);           //  8,388,608
    unsigned short* kb = (unsigned short*)(ws + 58867712);           //  8,388,608
    unsigned short* vt = (unsigned short*)(ws + 67256320);           //  8,388,608
    // total: 75,644,928 bytes

    int npairs = (MROWS * CC) / 2;                                   // 12,582,912
    cvt_x<<<npairs / 256, 256, 0, stream>>>(x, xb, npairs);
    cvt_w<<<(3 * WELEM + 255) / 256, 256, 0, stream>>>(Wq, Wk, Wv, wt);
    qkv_kernel<<<MROWS / 16 / 8, 256, 147456, stream>>>(xb, wt, qb, kb, vt);
    attn_kernel<<<(BB * TT / 16) / 8, 256, 139264, stream>>>(qb, kb, vt,
                                                             (float*)d_out);
}
